// GAT_61091614818586
// MI455X (gfx1250) — matrix operand order repaired and checked by reading, not compiled, not run on hardware
//
#include <hip/hip_runtime.h>
#include <hip/hip_bf16.h>

// ---------------------------------------------------------------------------
// GAT 2-layer forward for gfx1250 (wave32, WMMA bf16 path).
// N=10000 nodes, E=330000 edges, L1: 256->8x64, L2: 512->256.
// GEMMs via v_wmma_f32_16x16x32_bf16; edge softmax via encoded atomicMax +
// atomicAdd; aggregation wave-per-edge with L2-resident gathers + f32 atomics.
// ---------------------------------------------------------------------------

typedef __attribute__((ext_vector_type(16))) __bf16 v16bf;
typedef __attribute__((ext_vector_type(8)))  __bf16 v8bf;
typedef __attribute__((ext_vector_type(8)))  float  v8f;

#define HEADS 8
#define HID   64
#define INDIM 256
#define F1    (HEADS*HID)   // 512
#define OUTD  256
#define NEG_SLOPE 0.2f

// ---- monotone float<->uint encoding for atomicMax on floats -----------------
__device__ __forceinline__ unsigned f2mono(float f) {
  unsigned b = __float_as_uint(f);
  return (b & 0x80000000u) ? ~b : (b | 0x80000000u);
}
__device__ __forceinline__ float mono2f(unsigned u) {
  unsigned b = (u & 0x80000000u) ? (u ^ 0x80000000u) : ~u;
  return __uint_as_float(b);
}
#define ENC_NEG_INF 0x007FFFFFu   // f2mono(-inf)

// ---- fills ------------------------------------------------------------------
__global__ void k_fill_f32(float* p, float v, int n) {
  int i = blockIdx.x * blockDim.x + threadIdx.x;
  if (i < n) p[i] = v;
}
__global__ void k_fill_u32(unsigned* p, unsigned v, int n) {
  int i = blockIdx.x * blockDim.x + threadIdx.x;
  if (i < n) p[i] = v;
}

// ---- f32 -> bf16 conversions ------------------------------------------------
__global__ void k_cvt_bf16(const float* __restrict__ x, __bf16* __restrict__ y, int n) {
  int i = blockIdx.x * blockDim.x + threadIdx.x;
  if (i < n) y[i] = (__bf16)x[i];
}
// W[K][Nn] row-major -> Wt[Nn][K] bf16 (K contiguous per output row)
__global__ void k_transpose_cvt(const float* __restrict__ W, __bf16* __restrict__ Wt,
                                int K, int Nn) {
  int i = blockIdx.x * blockDim.x + threadIdx.x;
  if (i >= K * Nn) return;
  int n = i / K, k = i - n * K;
  Wt[i] = (__bf16)W[(size_t)k * Nn + n];
}

// ---- bf16 WMMA GEMM: C[M][N] = A[M][K] * Bt[N][K]^T -------------------------
// One 16x16 tile per wave; K stepped by 32 with v_wmma_f32_16x16x32_bf16.
// Fragment layouts per CDNA5 ISA (16-bit A 16x32, B 32x16, f32 C/D).
__global__ __launch_bounds__(256)
void k_wmma_gemm_bf16(const __bf16* __restrict__ A, const __bf16* __restrict__ Bt,
                      float* __restrict__ C, int M, int N, int K, int totalTiles) {
  int lane = threadIdx.x & 31;
  int wave = threadIdx.x >> 5;
  int tile = blockIdx.x * (blockDim.x >> 5) + wave;
  if (tile >= totalTiles) return;
  int ntN = N >> 4;
  int mt = tile / ntN, nt = tile - mt * ntN;

  int kh  = lane >> 4;      // which K-half this lane holds
  int l15 = lane & 15;
  const __bf16* arow = A  + (size_t)(mt * 16 + l15) * K;   // row M = l15
  const __bf16* bcol = Bt + (size_t)(nt * 16 + l15) * K;   // col N = l15

  v8f acc = {};
  for (int k0 = 0; k0 < K; k0 += 32) {
    // A lane(0-15): K = k0+[0..7], k0+16+[0..7]; lane(16-31): +8 shift
    v8bf a_lo = *(const v8bf*)(arow + k0 + kh * 8);
    v8bf a_hi = *(const v8bf*)(arow + k0 + 16 + kh * 8);
    // B lane(0-15): K = k0+[0..7], k0+16+[0..7]; lane(16-31): +8 shift
    v8bf b_lo = *(const v8bf*)(bcol + k0 + kh * 8);
    v8bf b_hi = *(const v8bf*)(bcol + k0 + 16 + kh * 8);
    v16bf a, b;
#pragma unroll
    for (int i = 0; i < 8; i++) {
      a[i] = a_lo[i]; a[i + 8] = a_hi[i];
      b[i] = b_lo[i]; b[i + 8] = b_hi[i];
    }
    acc = __builtin_amdgcn_wmma_f32_16x16x32_bf16(
        false, a, false, b, (short)0, acc, false, false);
  }
  // C/D: VGPR i -> M = i + 8*kh, N = l15
  float* cbase = C + (size_t)(mt * 16 + 8 * kh) * N + nt * 16 + l15;
#pragma unroll
  for (int i = 0; i < 8; i++) cbase[(size_t)i * N] = acc[i];
}

// ---- attention logits -------------------------------------------------------
// layer 1: el/er[n,h] = sum_d feat[n, h*64+d] * attn[h,d]
__global__ void k_el_er1(const float* __restrict__ feat, const float* __restrict__ al,
                         const float* __restrict__ ar, float* __restrict__ el,
                         float* __restrict__ er, int N) {
  int i = blockIdx.x * blockDim.x + threadIdx.x;
  if (i >= N * HEADS) return;
  int n = i >> 3, h = i & 7;
  const float* f = feat + (size_t)n * F1 + h * HID;
  const float* a = al + h * HID;
  const float* b = ar + h * HID;
  float sl = 0.f, sr = 0.f;
#pragma unroll 8
  for (int d = 0; d < HID; d++) { float v = f[d]; sl += v * a[d]; sr += v * b[d]; }
  el[i] = sl; er[i] = sr;
}
// layer 2 (H=1, D=256): wave-per-node reduction
__global__ void k_el_er2(const float* __restrict__ feat, const float* __restrict__ al,
                         const float* __restrict__ ar, float* __restrict__ el,
                         float* __restrict__ er, int N) {
  int wid  = (blockIdx.x * blockDim.x + threadIdx.x) >> 5;
  int lane = threadIdx.x & 31;
  if (wid >= N) return;
  const float* f = feat + (size_t)wid * OUTD;
  float sl = 0.f, sr = 0.f;
#pragma unroll
  for (int j = 0; j < 8; j++) {
    int idx = j * 32 + lane;
    float v = f[idx];
    sl += v * al[idx]; sr += v * ar[idx];
  }
#pragma unroll
  for (int o = 16; o > 0; o >>= 1) { sl += __shfl_xor(sl, o, 32); sr += __shfl_xor(sr, o, 32); }
  if (lane == 0) { el[wid] = sl; er[wid] = sr; }
}

// ---- edge softmax passes ----------------------------------------------------
template <int H>
__global__ void k_edge_logit_max(const int* __restrict__ src, const int* __restrict__ dst,
                                 const float* __restrict__ el, const float* __restrict__ er,
                                 float* __restrict__ ebuf, unsigned* __restrict__ menc, int E) {
  int i = blockIdx.x * blockDim.x + threadIdx.x;
  if (i >= E * H) return;
  int e = i / H, h = i - e * H;
  int s = src[e], d = dst[e];
  float v = el[s * H + h] + er[d * H + h];
  v = (v > 0.f) ? v : NEG_SLOPE * v;           // leaky_relu
  ebuf[i] = v;
  atomicMax(menc + d * H + h, f2mono(v));
}

template <int H>
__global__ void k_edge_exp_sum(const int* __restrict__ dst, float* __restrict__ ebuf,
                               const unsigned* __restrict__ menc,
                               float* __restrict__ denom, int E) {
  int i = blockIdx.x * blockDim.x + threadIdx.x;
  if (i >= E * H) return;
  int e = i / H, h = i - e * H;
  int d = dst[e];
  float m  = mono2f(menc[d * H + h]);
  float ex = __expf(ebuf[i] - m);
  ebuf[i] = ex;
  atomicAdd(denom + d * H + h, ex);
}

// ---- aggregation: out[dst] += alpha * feat[src] -----------------------------
// layer 1: 8 heads x 64 dims, wave per edge, 16 values/lane.
__global__ __launch_bounds__(256)
void k_edge_aggr1(const int* __restrict__ src, const int* __restrict__ dst,
                  const float* __restrict__ ebuf, const float* __restrict__ denom,
                  const float* __restrict__ feat, float* __restrict__ out, int E) {
  int wid  = blockIdx.x * (blockDim.x >> 5) + (threadIdx.x >> 5);
  int lane = threadIdx.x & 31;
  if (wid >= E) return;
  int s = src[wid], d = dst[wid];
  const float* fs = feat + (size_t)s * F1;
  float* od = out + (size_t)d * F1;
  __builtin_prefetch((const void*)(fs + lane * 16), 0, 0);  // global_prefetch_b8
  float alpha[HEADS];
#pragma unroll
  for (int h = 0; h < HEADS; h++) alpha[h] = ebuf[wid * HEADS + h] / denom[d * HEADS + h];
#pragma unroll
  for (int j = 0; j < 16; j++) {
    int idx = j * 32 + lane;           // head index = j>>1 (compile-time per j)
    atomicAdd(&od[idx], alpha[j >> 1] * fs[idx]);
  }
}
// layer 2: H=1, 256 dims, wave per edge, 8 values/lane.
__global__ __launch_bounds__(256)
void k_edge_aggr2(const int* __restrict__ src, const int* __restrict__ dst,
                  const float* __restrict__ ebuf, const float* __restrict__ denom,
                  const float* __restrict__ feat, float* __restrict__ out, int E) {
  int wid  = blockIdx.x * (blockDim.x >> 5) + (threadIdx.x >> 5);
  int lane = threadIdx.x & 31;
  if (wid >= E) return;
  int s = src[wid], d = dst[wid];
  const float* fs = feat + (size_t)s * OUTD;
  float* od = out + (size_t)d * OUTD;
  __builtin_prefetch((const void*)(fs + lane * 8), 0, 0);
  float alpha = ebuf[wid] / denom[d];
#pragma unroll
  for (int j = 0; j < 8; j++) {
    int idx = j * 32 + lane;
    atomicAdd(&od[idx], alpha * fs[idx]);
  }
}

// ---- ELU + downconvert for second GEMM input --------------------------------
__global__ void k_elu_bf16(const float* __restrict__ x, __bf16* __restrict__ y, int n) {
  int i = blockIdx.x * blockDim.x + threadIdx.x;
  if (i >= n) return;
  float v = x[i];
  v = (v > 0.f) ? v : (__expf(v) - 1.f);
  y[i] = (__bf16)v;
}

// ---------------------------------------------------------------------------
extern "C" void kernel_launch(void* const* d_in, const int* in_sizes, int n_in,
                              void* d_out, int out_size, void* d_ws, size_t ws_size,
                              hipStream_t stream) {
  const float* h   = (const float*)d_in[0];   // [N, 256]
  const float* W1  = (const float*)d_in[1];   // [256, 512]
  const float* al1 = (const float*)d_in[2];   // [8, 64]
  const float* ar1 = (const float*)d_in[3];   // [8, 64]
  const float* W2  = (const float*)d_in[4];   // [512, 256]
  const float* al2 = (const float*)d_in[5];   // [1, 256]
  const float* ar2 = (const float*)d_in[6];   // [1, 256]
  const int*   src = (const int*)d_in[7];     // [E]
  const int*   dst = (const int*)d_in[8];     // [E]
  float* out = (float*)d_out;                 // [N, 256]

  const int N = in_sizes[0] / INDIM;          // 10000
  const int E = in_sizes[7];                  // 330000

  // ---- workspace layout (256B aligned) ----
  size_t off = 0;
  char* base = (char*)d_ws;
  auto alloc = [&](size_t bytes) -> char* {
    char* p = base + off;
    off += (bytes + 255) & ~(size_t)255;
    return p;
  };
  __bf16*   hbf   = (__bf16*)  alloc((size_t)N * INDIM * 2);
  __bf16*   w1t   = (__bf16*)  alloc((size_t)F1 * INDIM * 2);   // [512][256]
  __bf16*   w2t   = (__bf16*)  alloc((size_t)OUTD * F1 * 2);    // [256][512]
  float*    feat1 = (float*)   alloc((size_t)N * F1 * 4);
  float*    el1   = (float*)   alloc((size_t)N * HEADS * 4);
  float*    er1   = (float*)   alloc((size_t)N * HEADS * 4);
  unsigned* menc1 = (unsigned*)alloc((size_t)N * HEADS * 4);
  float*    den1  = (float*)   alloc((size_t)N * HEADS * 4);
  float*    e1    = (float*)   alloc((size_t)E * HEADS * 4);
  float*    agg1  = (float*)   alloc((size_t)N * F1 * 4);
  __bf16*   xbf   = (__bf16*)  alloc((size_t)N * F1 * 2);
  float*    feat2 = (float*)   alloc((size_t)N * OUTD * 4);
  float*    el2   = (float*)   alloc((size_t)N * 4);
  float*    er2   = (float*)   alloc((size_t)N * 4);
  unsigned* menc2 = (unsigned*)alloc((size_t)N * 4);
  float*    den2  = (float*)   alloc((size_t)N * 4);
  float*    e2    = (float*)   alloc((size_t)E * 4);
  (void)ws_size; (void)n_in;

  const int T = 256;
  auto blks = [](int n, int t) { return (n + t - 1) / t; };

  // ---- precision conversion / weight transpose ----
  k_cvt_bf16<<<blks(N * INDIM, T), T, 0, stream>>>(h, hbf, N * INDIM);
  k_transpose_cvt<<<blks(INDIM * F1, T), T, 0, stream>>>(W1, w1t, INDIM, F1);
  k_transpose_cvt<<<blks(F1 * OUTD, T), T, 0, stream>>>(W2, w2t, F1, OUTD);

  // ---- layer 1: feat1 = h @ W1  (M=N, N=512, K=256) ----
  {
    int tiles = ((N + 15) / 16) * (F1 / 16);
    k_wmma_gemm_bf16<<<blks(tiles, 8), T, 0, stream>>>(hbf, w1t, feat1, N, F1, INDIM, tiles);
  }
  k_el_er1<<<blks(N * HEADS, T), T, 0, stream>>>(feat1, al1, ar1, el1, er1, N);

  k_fill_u32<<<blks(N * HEADS, T), T, 0, stream>>>(menc1, ENC_NEG_INF, N * HEADS);
  k_fill_f32<<<blks(N * HEADS, T), T, 0, stream>>>(den1, 0.f, N * HEADS);
  k_fill_f32<<<blks(N * F1, T), T, 0, stream>>>(agg1, 0.f, N * F1);

  k_edge_logit_max<HEADS><<<blks(E * HEADS, T), T, 0, stream>>>(src, dst, el1, er1, e1, menc1, E);
  k_edge_exp_sum<HEADS><<<blks(E * HEADS, T), T, 0, stream>>>(dst, e1, menc1, den1, E);
  k_edge_aggr1<<<blks(E, 8), T, 0, stream>>>(src, dst, e1, den1, feat1, agg1, E);

  // ---- ELU + bf16 downconvert ----
  k_elu_bf16<<<blks(N * F1, T), T, 0, stream>>>(agg1, xbf, N * F1);

  // ---- layer 2: feat2 = x @ W2  (M=N, N=256, K=512) ----
  {
    int tiles = ((N + 15) / 16) * (OUTD / 16);
    k_wmma_gemm_bf16<<<blks(tiles, 8), T, 0, stream>>>(xbf, w2t, feat2, N, OUTD, F1, tiles);
  }
  k_el_er2<<<blks(N * 32, T), T, 0, stream>>>(feat2, al2, ar2, el2, er2, N);

  k_fill_u32<<<blks(N, T), T, 0, stream>>>(menc2, ENC_NEG_INF, N);
  k_fill_f32<<<blks(N, T), T, 0, stream>>>(den2, 0.f, N);
  k_fill_f32<<<blks(out_size, T), T, 0, stream>>>(out, 0.f, out_size);

  k_edge_logit_max<1><<<blks(E, T), T, 0, stream>>>(src, dst, el2, er2, e2, menc2, E);
  k_edge_exp_sum<1><<<blks(E, T), T, 0, stream>>>(dst, e2, menc2, den2, E);
  k_edge_aggr2<<<blks(E, 8), T, 0, stream>>>(src, dst, e2, den2, feat2, out, E);
}
